// InceptionL_16166256902763
// MI455X (gfx1250) — matrix operand order repaired and checked by reading, not compiled, not run on hardware
//
#include <hip/hip_runtime.h>
#include <math.h>

typedef __bf16 bf16_t;
typedef __attribute__((ext_vector_type(16))) __bf16 v16bf;
typedef __attribute__((ext_vector_type(8)))  __bf16 v8bf;
typedef __attribute__((ext_vector_type(8)))  float  v8f;

// ---------------- degree / norm ----------------
__global__ void k_deg_count(const int* __restrict__ dst, int* __restrict__ deg, int E) {
    int t = blockIdx.x * blockDim.x + threadIdx.x;
    if (t < E) atomicAdd(&deg[dst[t]], 1);
}
__global__ void k_dis(const int* __restrict__ deg, float* __restrict__ dis, int N) {
    int t = blockIdx.x * blockDim.x + threadIdx.x;
    if (t < N) dis[t] = rsqrtf((float)deg[t] + 1.0f);
}

// ---------------- conversions ----------------
__global__ void k_cvt_bf16(const float* __restrict__ in, bf16_t* __restrict__ out, int n) {
    int t = blockIdx.x * blockDim.x + threadIdx.x;
    if (t < n) out[t] = (bf16_t)in[t];
}
// W [K,F] f32 row-major -> Wt [F,K] bf16 (column-major for contiguous B fragments)
__global__ void k_cvt_w_t(const float* __restrict__ W, bf16_t* __restrict__ Wt, int K, int F) {
    int t = blockIdx.x * blockDim.x + threadIdx.x;
    if (t < K * F) { int k = t / F, f = t - k * F; Wt[f * K + k] = (bf16_t)W[t]; }
}

// ---------------- WMMA GEMM: C[M,F] = A[M,K](bf16) * B[K,F](bf16, stored transposed) ----
// Each wave: one 16x64 tile. acc init: accumulate ? C : (addrow ? addrow : 0).
__global__ __launch_bounds__(256)
void k_gemm_bf16(const bf16_t* __restrict__ A, const bf16_t* __restrict__ Bt,
                 float* __restrict__ C, const float* __restrict__ addrow,
                 int M, int K, int F, int ldbt, int bk0, int accumulate)
{
    int wave = (blockIdx.x * blockDim.x + threadIdx.x) >> 5;
    int lane = threadIdx.x & 31;
    int colTiles = F >> 6;
    int rowTiles = M >> 4;
    if (wave >= rowTiles * colTiles) return;          // wave-uniform: EXEC all-1 for WMMA
    int rt   = wave / colTiles;
    int col0 = (wave - rt * colTiles) << 6;
    int half = lane >> 4;                              // 0/1 (K-interleave half)
    int l16  = lane & 15;

    v8f acc[4];
    if (accumulate) {
        #pragma unroll
        for (int j = 0; j < 4; ++j)
            #pragma unroll
            for (int v = 0; v < 8; ++v)
                acc[j][v] = C[(long)((rt << 4) + half * 8 + v) * F + col0 + j * 16 + l16];
    } else if (addrow) {
        #pragma unroll
        for (int j = 0; j < 4; ++j) {
            float av = addrow[col0 + j * 16 + l16];
            #pragma unroll
            for (int v = 0; v < 8; ++v) acc[j][v] = av;
        }
    } else {
        #pragma unroll
        for (int j = 0; j < 4; ++j) acc[j] = (v8f){0.f,0.f,0.f,0.f,0.f,0.f,0.f,0.f};
    }

    // A: lane holds row m = rt*16 + l16; elems 0..15 -> K = half*16 + 0..15
    const bf16_t* arow  = A + (long)((rt << 4) + l16) * K + half * 16;
    // B (transposed): lane holds col n; elems 0..15 -> K = half*16 + 0..15
    const bf16_t* bbase = Bt + bk0 + half * 16;

    for (int kk = 0; kk < K; kk += 32) {
        v8bf alo = *(const v8bf*)(arow + kk);
        v8bf ahi = *(const v8bf*)(arow + kk + 8);
        v16bf afrag = __builtin_shufflevector(alo, ahi, 0,1,2,3,4,5,6,7,8,9,10,11,12,13,14,15);
        #pragma unroll
        for (int j = 0; j < 4; ++j) {
            int n = col0 + j * 16 + l16;
            v16bf bfrag = *(const v16bf*)(bbase + (long)n * ldbt + kk);
            acc[j] = __builtin_amdgcn_wmma_f32_16x16x32_bf16(
                         false, afrag, false, bfrag, (short)0, acc[j], false, false);
        }
    }
    #pragma unroll
    for (int j = 0; j < 4; ++j)
        #pragma unroll
        for (int v = 0; v < 8; ++v)
            C[(long)((rt << 4) + half * 8 + v) * F + col0 + j * 16 + l16] = acc[j][v];
}

// ---------------- GCN aggregation ----------------
__global__ void k_init_agg(const float* __restrict__ H, const float* __restrict__ dis,
                           float* __restrict__ agg, int N, int F) {
    int t = blockIdx.x * blockDim.x + threadIdx.x;
    if (t < N * F) { float d = dis[t / F]; agg[t] = H[t] * d * d; }
}
__global__ void k_edge_scatter(const int* __restrict__ src, const int* __restrict__ dst,
                               const float* __restrict__ dis, const float* __restrict__ H,
                               float* __restrict__ agg, int E, int F) {
    int f4 = F >> 2;
    int t = blockIdx.x * blockDim.x + threadIdx.x;
    if (t >= E * f4) return;
    int e = t / f4;
    int c = (t - e * f4) << 2;
    int s = src[e], d = dst[e];
    float nrm = dis[s] * dis[d];
    const float4 h = *(const float4*)(H + (long)s * F + c);
    float* o = agg + (long)d * F + c;
    atomicAdd(o + 0, h.x * nrm);
    atomicAdd(o + 1, h.y * nrm);
    atomicAdd(o + 2, h.z * nrm);
    atomicAdd(o + 3, h.w * nrm);
}
__global__ void k_act_bf16(const float* __restrict__ agg, const float* __restrict__ b,
                           bf16_t* __restrict__ out, int N, int F) {
    int t = blockIdx.x * blockDim.x + threadIdx.x;
    if (t < N * F) out[t] = (bf16_t)tanhf(agg[t] + b[t % F]);
}
__global__ void k_act_f32(const float* __restrict__ agg, const float* __restrict__ b,
                          float* __restrict__ out, int N, int F) {
    int t = blockIdx.x * blockDim.x + threadIdx.x;
    if (t < N * F) out[t] = tanhf(agg[t] + b[t % F]);
}

// ---------------- pooling (global column max) + constant row ----------------
__global__ void k_colmax_part(const float* __restrict__ T, float* __restrict__ part,
                              int N, int F, int rowsPerBlock) {
    int col = threadIdx.x;                 // blockDim.x == F
    int r0 = blockIdx.x * rowsPerBlock;
    int r1 = min(N, r0 + rowsPerBlock);
    float m = -INFINITY;
    for (int r = r0; r < r1; ++r) m = fmaxf(m, T[(long)r * F + col]);
    part[blockIdx.x * F + col] = m;
}
__global__ void k_colmax_final(const float* __restrict__ part, float* __restrict__ pooled,
                               int nb, int F) {
    int col = threadIdx.x;
    float m = -INFINITY;
    for (int b = 0; b < nb; ++b) m = fmaxf(m, part[b * F + col]);
    pooled[col] = m;
}
// crow[f] = sum_k pooled[k] * W7[k*F+f], k in [0,Kp)
__global__ void k_constrow(const float* __restrict__ pooled, const float* __restrict__ W7,
                           float* __restrict__ crow, int Kp, int F) {
    int f = threadIdx.x;
    if (f < F) {
        float s = 0.f;
        for (int k = 0; k < Kp; ++k) s += pooled[k] * W7[k * F + f];
        crow[f] = s;
    }
}

extern "C" void kernel_launch(void* const* d_in, const int* in_sizes, int n_in,
                              void* d_out, int out_size, void* d_ws, size_t ws_size,
                              hipStream_t stream) {
    const float* x   = (const float*)d_in[0];
    const int*   ei  = (const int*)  d_in[1];
    const float* W1  = (const float*)d_in[3];  const float* b1 = (const float*)d_in[4];
    const float* W2  = (const float*)d_in[5];  const float* b2 = (const float*)d_in[6];
    const float* W3  = (const float*)d_in[7];  const float* b3 = (const float*)d_in[8];
    const float* W4  = (const float*)d_in[9];  const float* b4 = (const float*)d_in[10];
    const float* W7  = (const float*)d_in[11]; const float* b7 = (const float*)d_in[12];

    const int N = in_sizes[0] / 64;        // 50000 (multiple of 16)
    const int E = in_sizes[1] / 2;         // 400000
    const int* src = ei;
    const int* dst = ei + E;
    const int D = 64;

    // ---- workspace layout (256B aligned) ----
    char* base = (char*)d_ws;  size_t off = 0;
    auto alloc = [&](size_t bytes) { void* p = base + off; off = (off + bytes + 255) & ~(size_t)255; return p; };
    int*    deg    = (int*)   alloc((size_t)N * 4);
    float*  dis    = (float*) alloc((size_t)N * 4);
    bf16_t* Xbf    = (bf16_t*)alloc((size_t)N * D * 2);
    bf16_t* W1t    = (bf16_t*)alloc((size_t)64  * 128 * 2);
    bf16_t* W2t    = (bf16_t*)alloc((size_t)128 * 256 * 2);
    bf16_t* W3t    = (bf16_t*)alloc((size_t)64  * 128 * 2);
    bf16_t* W4t    = (bf16_t*)alloc((size_t)128 * 256 * 2);
    bf16_t* W7t    = (bf16_t*)alloc((size_t)576 * 128 * 2);   // [F=128, K=576]
    float*  H      = (float*) alloc((size_t)N * 256 * 4);
    float*  AGG    = (float*) alloc((size_t)N * 256 * 4);
    bf16_t* T128   = (bf16_t*)alloc((size_t)N * 128 * 2);
    bf16_t* T256   = (bf16_t*)alloc((size_t)N * 256 * 2);
    const int nbMax = (N + 255) / 256;
    float*  part   = (float*) alloc((size_t)nbMax * 256 * 4);
    float*  pooled = (float*) alloc(256 * 4);
    float*  crow   = (float*) alloc(128 * 4);

    const int B = 256;
    auto g = [](long n, int b) { return (int)((n + b - 1) / b); };

    // ---- degrees & normalization ----
    hipMemsetAsync(deg, 0, (size_t)N * 4, stream);
    k_deg_count<<<g(E, B), B, 0, stream>>>(dst, deg, E);
    k_dis<<<g(N, B), B, 0, stream>>>(deg, dis, N);

    // ---- precision conversions ----
    k_cvt_bf16<<<g((long)N * D, B), B, 0, stream>>>(x, Xbf, N * D);
    k_cvt_w_t<<<g(64 * 128, B), B, 0, stream>>>(W1, W1t, 64, 128);
    k_cvt_w_t<<<g(128 * 256, B), B, 0, stream>>>(W2, W2t, 128, 256);
    k_cvt_w_t<<<g(64 * 128, B), B, 0, stream>>>(W3, W3t, 64, 128);
    k_cvt_w_t<<<g(128 * 256, B), B, 0, stream>>>(W4, W4t, 128, 256);
    k_cvt_w_t<<<g(576 * 128, B), B, 0, stream>>>(W7, W7t, 576, 128);

    auto gemm = [&](const bf16_t* A, const bf16_t* Bt, float* C, const float* addrow,
                    int K, int F, int ldbt, int bk0, int acc) {
        int tiles = (N / 16) * (F / 64);
        k_gemm_bf16<<<g(tiles, 8), 256, 0, stream>>>(A, Bt, C, addrow, N, K, F, ldbt, bk0, acc);
    };
    auto aggregate = [&](int F) {
        k_init_agg<<<g((long)N * F, B), B, 0, stream>>>(H, dis, AGG, N, F);
        k_edge_scatter<<<g((long)E * (F / 4), B), B, 0, stream>>>(src, dst, dis, H, AGG, E, F);
    };

    // ---- conv1: tanh(gcn(x, W1)) -> T128 (bf16) ----
    gemm(Xbf, W1t, H, nullptr, 64, 128, 64, 0, 0);
    aggregate(128);
    k_act_bf16<<<g((long)N * 128, B), B, 0, stream>>>(AGG, b1, T128, N, 128);

    // ---- conv2: tanh(gcn(T128, W2)) -> H (f32, for pooling only) ----
    gemm(T128, W2t, H, nullptr, 128, 256, 128, 0, 0);
    aggregate(256);
    k_act_f32<<<g((long)N * 256, B), B, 0, stream>>>(AGG, b2, H, N, 256);

    // ---- pooling: global column max over all nodes; const row = pooled @ W7[0:256,:] ----
    int nb = (N + 255) / 256;
    k_colmax_part<<<nb, 256, 0, stream>>>(H, part, N, 256, 256);
    k_colmax_final<<<1, 256, 0, stream>>>(part, pooled, nb, 256);
    k_constrow<<<1, 128, 0, stream>>>(pooled, W7, crow, 256, 128);

    // ---- conv3: tanh(gcn(x, W3)) -> T128 ----
    gemm(Xbf, W3t, H, nullptr, 64, 128, 64, 0, 0);
    aggregate(128);
    k_act_bf16<<<g((long)N * 128, B), B, 0, stream>>>(AGG, b3, T128, N, 128);

    // ---- conv4: tanh(gcn(T128, W4)) -> T256 ----
    gemm(T128, W4t, H, nullptr, 128, 256, 128, 0, 0);
    aggregate(256);
    k_act_bf16<<<g((long)N * 256, B), B, 0, stream>>>(AGG, b4, T256, N, 256);

    // ---- conv7: h = crow + T256 @ W7[256:512] + x @ W7[512:576]; out = tanh(gcn agg + b7)
    gemm(T256, W7t, H, crow, 256, 128, 576, 256, 0);
    gemm(Xbf,  W7t, H, nullptr, 64, 128, 576, 512, 1);
    aggregate(128);
    k_act_f32<<<g((long)N * 128, B), B, 0, stream>>>(AGG, b7, (float*)d_out, N, 128);
}
